// MultiHeadAttention_3693671875079
// MI455X (gfx1250) — matrix operand order repaired and checked by reading, not compiled, not run on hardware
//
#include <hip/hip_runtime.h>
#include <cstdint>

// Problem constants (match reference)
#define S_LEN   2048
#define DMODEL  1024
#define NHEAD   16
#define DK      64
#define BATCH   2

typedef __attribute__((ext_vector_type(16))) __bf16 v16bf;
typedef __attribute__((ext_vector_type(8)))  __bf16 v8bf;
typedef __attribute__((ext_vector_type(8)))  float  v8f;
typedef __attribute__((ext_vector_type(4)))  unsigned int v4u;
typedef __attribute__((ext_vector_type(8)))  int v8i;
typedef __attribute__((ext_vector_type(4)))  int v4i;

union FragU { v16bf v; v8bf h[2]; __bf16 e[16]; };

// ---- WMMA fragment loaders (wave32; A and B both use contiguous k = hi*16 + e, not the ISA table order) ----
// A 16x32 bf16 tile, row-major, K-contiguous. Per lane (r=lane&15, hi=lane>>4):
//   e0..e15 = A[r, hi*16 + 0..15]
__device__ __forceinline__ v16bf load_frag_a16(const __bf16* tile, int ld, int lane) {
  const int r = lane & 15, hi = (lane >> 4) & 1;
  const __bf16* p = tile + (size_t)r * ld + hi * 16;
  FragU f;
  f.h[0] = *(const v8bf*)(p);
  f.h[1] = *(const v8bf*)(p + 8);
  return f.v;
}
// B 32x16 bf16 tile given as Bt (N x K row-major). Per lane: column n0+r,
//   e0..e15 = Bt[r, hi*16 + 0..15] (16 contiguous K values)
__device__ __forceinline__ v16bf load_frag_b16(const __bf16* tile, int ld, int lane) {
  const int r = lane & 15, hi = (lane >> 4) & 1;
  const __bf16* p = tile + (size_t)r * ld + hi * 16;
  FragU f;
  f.h[0] = *(const v8bf*)(p);
  f.h[1] = *(const v8bf*)(p + 8);
  return f.v;
}
// B fragment from an LDS tile with padded row stride of 40 elements (80 bytes):
// TDM pads 16B per 64B row so rows r=0..15 land on distinct LDS banks.
__device__ __forceinline__ v16bf load_frag_b_lds(const __bf16* tile, int lane) {
  const int r = lane & 15, hi = (lane >> 4) & 1;
  const __bf16* p = tile + r * 40 + hi * 16;
  FragU f;
  f.h[0] = *(const v8bf*)(p);
  f.h[1] = *(const v8bf*)(p + 8);
  return f.v;
}
// A fragment from an fp32 row-major matrix, converted to bf16 in registers.
__device__ __forceinline__ v16bf load_frag_a_f32(const float* tile, int ld, int lane) {
  const int r = lane & 15, hi = (lane >> 4) & 1;
  const float* p = tile + (size_t)r * ld + hi * 16;
  v8f x0 = *(const v8f*)(p);
  v8f x1 = *(const v8f*)(p + 8);
  FragU f;
#pragma unroll
  for (int j = 0; j < 8; ++j) { f.e[j] = (__bf16)x0[j]; f.e[8 + j] = (__bf16)x1[j]; }
  return f.v;
}

__device__ __forceinline__ v8f wmma_bf16(v16bf a, v16bf b, v8f c) {
  // 8 args: (neg_a, A, neg_b, B, c_mod, C, reuse_a, reuse_b)
  return __builtin_amdgcn_wmma_f32_16x16x32_bf16(false, a, false, b, (short)0, c, false, false);
}

// ---- Tensor Data Mover: load a 128(row) x 32(bf16) weight tile into LDS ----
// D# built per CDNA5 ISA. Padding: after every 64B row,
// insert 16B -> LDS row stride 80B (bank-conflict-free b128 fragment reads).
__device__ __forceinline__ void tdm_load_b_tile(unsigned long long gbase,
                                                int k0, unsigned ldsAddr) {
  const unsigned long long ga = gbase + (unsigned long long)k0 * 2ull;
  v4u g0;
  g0.x = 1u;                                                   // count=1 (valid user D#)
  g0.y = ldsAddr;                                              // lds_addr (bytes)
  g0.z = (unsigned)ga;                                         // global_addr[31:0]
  g0.w = (unsigned)((ga >> 32) & 0x1FFFFFFull) | (2u << 30);   // addr[56:32] | type=2
  v8i g1;
  g1[0] = (int)((1u << 16)      // data_size = 2 bytes
              | (1u << 20)      // pad_enable
              | (3u << 22)      // pad_interval code 3 = 16 DWORDs (64B)
              | (3u << 25));    // pad_amount  code 3 = 4 DWORDs (16B)
  g1[1] = (int)(1024u << 16);   // tensor_dim0 = 1024 (bits 79:48, low half)
  g1[2] = (int)(1024u << 16);   // tensor_dim0 hi | tensor_dim1 = 1024 (low half)
  g1[3] = (int)(32u << 16);     // tensor_dim1 hi | tile_dim0 = 32
  g1[4] = 128;                  // tile_dim1 = 128 | tile_dim2 = 0
  g1[5] = 1024;                 // tensor_dim0_stride[31:0] = 1024 elements
  g1[6] = 0;                    // stride hi | tensor_dim1_stride lo (unused, 2D)
  g1[7] = 0;
  v4i g2 = {0, 0, 0, 0}, g3 = {0, 0, 0, 0};
  v8i g4 = {0, 0, 0, 0, 0, 0, 0, 0};     // extra group (clang-23 6-arg form)
  __builtin_amdgcn_tensor_load_to_lds(g0, g1, g2, g3, g4, 0);
}

// ---------------- fp32 -> bf16 conversion ----------------
__global__ void __launch_bounds__(256)
cvt_f32_bf16(const float* __restrict__ s, __bf16* __restrict__ d, int n) {
  for (int i = blockIdx.x * 256 + threadIdx.x; i < n; i += gridDim.x * 256)
    d[i] = (__bf16)s[i];
}

// ---------------- Projection GEMM: P = X(4096x1024) @ W^T + bias ----------------
// Weight tiles staged LDS-side by the Tensor Data Mover (double-buffered).
// mode 0: bf16 dst, head layout (b,h,s,dk), val=(acc+bias)*scale   (Q: scale=1/8, K: 1)
// mode 1: bf16 dst, transposed head layout (b,h,dk,s)              (V^T)
// mode 2: fp32 dst, row-major (B*S, D)                             (final out @ Wo^T + bo)
__global__ void __launch_bounds__(256)
proj_gemm(const __bf16* __restrict__ X, const __bf16* __restrict__ W,
          const float* __restrict__ bias, __bf16* __restrict__ dst_bf,
          float* __restrict__ dst_f, int mode, float scale) {
  __shared__ __bf16 ldsB[2][128 * 40];   // 128 rows x 32 elems, 80B padded stride
  const int lane = threadIdx.x & 31;
  const int wid  = threadIdx.x >> 5;
  const int m0   = blockIdx.y * 64 + (wid >> 2) * 32;
  const int nBlk = blockIdx.x * 128;
  const int nLoc = (wid & 3) * 32;
  const int r = lane & 15, hi = (lane >> 4) & 1;

  const unsigned ldsAddr0 = (unsigned)(uintptr_t)&ldsB[0][0];
  const unsigned ldsAddr1 = (unsigned)(uintptr_t)&ldsB[1][0];
  const unsigned long long gW =
      (unsigned long long)(uintptr_t)(W + (size_t)nBlk * DMODEL);

  if (wid == 0) {                         // prologue: fill both buffers
    tdm_load_b_tile(gW, 0,  ldsAddr0);
    tdm_load_b_tile(gW, 32, ldsAddr1);
  }

  v8f acc[2][2] = {};
  const __bf16* Ap0 = X + (size_t)m0 * DMODEL;
  const __bf16* Ap1 = X + (size_t)(m0 + 16) * DMODEL;

  auto kstep = [&](int k0, int cur) {
    __builtin_prefetch(Ap0 + k0 + 128, 0, 0);  // global_prefetch_b8
    v16bf a0 = load_frag_a16(Ap0 + k0, DMODEL, lane);
    v16bf a1 = load_frag_a16(Ap1 + k0, DMODEL, lane);
    v16bf b0 = load_frag_b_lds(&ldsB[cur][nLoc * 40], lane);
    v16bf b1 = load_frag_b_lds(&ldsB[cur][(nLoc + 16) * 40], lane);
    acc[0][0] = wmma_bf16(a0, b0, acc[0][0]);
    acc[0][1] = wmma_bf16(a0, b1, acc[0][1]);
    acc[1][0] = wmma_bf16(a1, b0, acc[1][0]);
    acc[1][1] = wmma_bf16(a1, b1, acc[1][1]);
  };

  // Steady state: wait for the oldest of (at most) two in-flight TDM loads.
  // TENSORcnt is per-wave: waves that never issue TDM pass the wait for free,
  // so no divergent guard is needed around it.
  int k0 = 0;
  for (; k0 < DMODEL - 32; k0 += 32) {
    const int cur = (k0 >> 5) & 1;
    __builtin_amdgcn_s_wait_tensorcnt(1);
    __syncthreads();                      // `cur` tile visible to all waves
    kstep(k0, cur);
    __syncthreads();                      // everyone done reading `cur`
    if (wid == 0 && k0 + 64 < DMODEL)     // refill the buffer just consumed
      tdm_load_b_tile(gW, k0 + 64, cur ? ldsAddr1 : ldsAddr0);
  }
  // Peeled last iteration: drain the TDM pipe completely.
  __builtin_amdgcn_s_wait_tensorcnt(0);
  __syncthreads();
  kstep(k0, (k0 >> 5) & 1);

#pragma unroll
  for (int im = 0; im < 2; ++im)
#pragma unroll
    for (int in = 0; in < 2; ++in)
#pragma unroll
      for (int i = 0; i < 8; ++i) {
        const int m = m0 + im * 16 + hi * 8 + i;
        const int n = nBlk + nLoc + in * 16 + r;
        const float val = (acc[im][in][i] + bias[n]) * scale;
        const int b = m >> 11, s = m & (S_LEN - 1);
        const int h = n >> 6,  dk = n & (DK - 1);
        if (mode == 0)
          dst_bf[((size_t)(b * NHEAD + h) * S_LEN + s) * DK + dk] = (__bf16)val;
        else if (mode == 1)
          dst_bf[((size_t)(b * NHEAD + h) * DK + dk) * S_LEN + s] = (__bf16)val;
        else
          dst_f[(size_t)m * DMODEL + n] = val;
      }
}

// ---------------- Scores: logits = Qh(2048x64) @ Kh^T, per (b,h) ----------------
__global__ void __launch_bounds__(256)
scores_gemm(const __bf16* __restrict__ Qh, const __bf16* __restrict__ Kh,
            float* __restrict__ attn) {
  const int bh = blockIdx.z;
  const __bf16* Q = Qh + (size_t)bh * S_LEN * DK;
  const __bf16* K = Kh + (size_t)bh * S_LEN * DK;
  float* Ao = attn + (size_t)bh * S_LEN * S_LEN;

  const int lane = threadIdx.x & 31;
  const int wid  = threadIdx.x >> 5;
  const int m0 = blockIdx.y * 64  + (wid >> 2) * 32;
  const int n0 = blockIdx.x * 128 + (wid & 3)  * 32;
  const int r = lane & 15, hi = (lane >> 4) & 1;

  v8f acc[2][2] = {};
#pragma unroll
  for (int k0 = 0; k0 < DK; k0 += 32) {
    v16bf a0 = load_frag_a16(Q + (size_t)m0 * DK + k0, DK, lane);
    v16bf a1 = load_frag_a16(Q + (size_t)(m0 + 16) * DK + k0, DK, lane);
    v16bf b0 = load_frag_b16(K + (size_t)n0 * DK + k0, DK, lane);
    v16bf b1 = load_frag_b16(K + (size_t)(n0 + 16) * DK + k0, DK, lane);
    acc[0][0] = wmma_bf16(a0, b0, acc[0][0]);
    acc[0][1] = wmma_bf16(a0, b1, acc[0][1]);
    acc[1][0] = wmma_bf16(a1, b0, acc[1][0]);
    acc[1][1] = wmma_bf16(a1, b1, acc[1][1]);
  }

#pragma unroll
  for (int im = 0; im < 2; ++im)
#pragma unroll
    for (int in = 0; in < 2; ++in)
#pragma unroll
      for (int i = 0; i < 8; ++i) {
        const int m = m0 + im * 16 + hi * 8 + i;
        const int n = n0 + in * 16 + r;
        Ao[(size_t)m * S_LEN + n] = acc[im][in][i];
      }
}

// ---------------- Row softmax with key mask (in place, fp32) ----------------
__global__ void __launch_bounds__(256)
softmax_rows(float* __restrict__ attn, const int* __restrict__ mask) {
  const int row = blockIdx.x;                 // (b*H + h)*S + m
  const int b = row >> 15;                    // H*S = 32768
  float* p = attn + (size_t)row * S_LEN;
  const int* mrow = mask + b * S_LEN;

  __shared__ float red[256];
  const int tid = threadIdx.x;
  float vals[8];
  float lmax = -3.4e38f;
#pragma unroll
  for (int j = 0; j < 8; ++j) {
    const int c = tid + j * 256;
    float x = p[c];
    if (mrow[c] == 0) x = -1.0e9f;
    vals[j] = x;
    lmax = fmaxf(lmax, x);
  }
  red[tid] = lmax; __syncthreads();
  for (int s = 128; s > 0; s >>= 1) {
    if (tid < s) red[tid] = fmaxf(red[tid], red[tid + s]);
    __syncthreads();
  }
  const float rmax = red[0]; __syncthreads();

  float lsum = 0.f;
#pragma unroll
  for (int j = 0; j < 8; ++j) { vals[j] = __expf(vals[j] - rmax); lsum += vals[j]; }
  red[tid] = lsum; __syncthreads();
  for (int s = 128; s > 0; s >>= 1) {
    if (tid < s) red[tid] += red[tid + s];
    __syncthreads();
  }
  const float inv = 1.0f / red[0];
#pragma unroll
  for (int j = 0; j < 8; ++j) p[tid + j * 256] = vals[j] * inv;
}

// ---------------- attn(2048x2048,f32) @ V  (via V^T, NT form) ----------------
__global__ void __launch_bounds__(256)
av_gemm(const float* __restrict__ attn, const __bf16* __restrict__ Vt,
        __bf16* __restrict__ ctx) {
  const int bh = blockIdx.y;
  const int b = bh >> 4, h = bh & 15;
  const float* A  = attn + (size_t)bh * S_LEN * S_LEN;
  const __bf16* Bt = Vt + (size_t)bh * DK * S_LEN;   // (DK rows) x (S cols), row-major

  const int lane = threadIdx.x & 31;
  const int wid  = threadIdx.x >> 5;
  const int m0 = blockIdx.x * 128 + (wid >> 1) * 32; // query index
  const int n0 = (wid & 1) * 32;                     // dk index
  const int r = lane & 15, hi = (lane >> 4) & 1;

  v8f acc[2][2] = {};
#pragma unroll 2
  for (int k0 = 0; k0 < S_LEN; k0 += 32) {
    v16bf a0 = load_frag_a_f32(A + (size_t)m0 * S_LEN + k0, S_LEN, lane);
    v16bf a1 = load_frag_a_f32(A + (size_t)(m0 + 16) * S_LEN + k0, S_LEN, lane);
    v16bf b0 = load_frag_b16(Bt + (size_t)n0 * S_LEN + k0, S_LEN, lane);
    v16bf b1 = load_frag_b16(Bt + (size_t)(n0 + 16) * S_LEN + k0, S_LEN, lane);
    acc[0][0] = wmma_bf16(a0, b0, acc[0][0]);
    acc[0][1] = wmma_bf16(a0, b1, acc[0][1]);
    acc[1][0] = wmma_bf16(a1, b0, acc[1][0]);
    acc[1][1] = wmma_bf16(a1, b1, acc[1][1]);
  }

#pragma unroll
  for (int im = 0; im < 2; ++im)
#pragma unroll
    for (int in = 0; in < 2; ++in)
#pragma unroll
      for (int i = 0; i < 8; ++i) {
        const int m = m0 + im * 16 + hi * 8 + i;      // s
        const int n = n0 + in * 16 + r;               // dk
        ctx[((size_t)(b * S_LEN + m)) * DMODEL + h * DK + n] = (__bf16)acc[im][in][i];
      }
}

// ---------------- host-side orchestration ----------------
extern "C" void kernel_launch(void* const* d_in, const int* in_sizes, int n_in,
                              void* d_out, int out_size, void* d_ws, size_t ws_size,
                              hipStream_t stream) {
  (void)in_sizes; (void)n_in; (void)out_size; (void)ws_size;
  const float* q  = (const float*)d_in[0];
  const float* k  = (const float*)d_in[1];
  const float* v  = (const float*)d_in[2];
  const int*  msk = (const int*)  d_in[3];
  const float* Wq = (const float*)d_in[4];
  const float* bq = (const float*)d_in[5];
  const float* Wk = (const float*)d_in[6];
  const float* bk = (const float*)d_in[7];
  const float* Wv = (const float*)d_in[8];
  const float* bv = (const float*)d_in[9];
  const float* Wo = (const float*)d_in[10];
  const float* bo = (const float*)d_in[11];

  float* out  = (float*)d_out;                                  // B*S*D
  float* attn = out + (size_t)BATCH * S_LEN * DMODEL;           // B*H*S*S

  const size_t BSD = (size_t)BATCH * S_LEN * DMODEL;            // 4,194,304
  const size_t DD  = (size_t)DMODEL * DMODEL;                   // 1,048,576
  __bf16* wsp = (__bf16*)d_ws;                                  // ~64 MiB total
  __bf16* qb  = wsp;        __bf16* kb  = qb  + BSD;  __bf16* vb  = kb  + BSD;
  __bf16* wqb = vb  + BSD;  __bf16* wkb = wqb + DD;   __bf16* wvb = wkb + DD;
  __bf16* wob = wvb + DD;
  __bf16* qh  = wob + DD;   __bf16* kh  = qh  + BSD;  __bf16* vt  = kh  + BSD;
  __bf16* ctx = vt  + BSD;

  const dim3 blk(256);

  cvt_f32_bf16<<<dim3(2048), blk, 0, stream>>>(q, qb, (int)BSD);
  cvt_f32_bf16<<<dim3(2048), blk, 0, stream>>>(k, kb, (int)BSD);
  cvt_f32_bf16<<<dim3(2048), blk, 0, stream>>>(v, vb, (int)BSD);
  cvt_f32_bf16<<<dim3(1024), blk, 0, stream>>>(Wq, wqb, (int)DD);
  cvt_f32_bf16<<<dim3(1024), blk, 0, stream>>>(Wk, wkb, (int)DD);
  cvt_f32_bf16<<<dim3(1024), blk, 0, stream>>>(Wv, wvb, (int)DD);
  cvt_f32_bf16<<<dim3(1024), blk, 0, stream>>>(Wo, wob, (int)DD);

  const dim3 gProj(DMODEL / 128, (BATCH * S_LEN) / 64);         // (8, 64)
  proj_gemm<<<gProj, blk, 0, stream>>>(qb, wqb, bq, qh, nullptr, 0, 0.125f); // fold 1/sqrt(Dk)
  proj_gemm<<<gProj, blk, 0, stream>>>(kb, wkb, bk, kh, nullptr, 0, 1.0f);
  proj_gemm<<<gProj, blk, 0, stream>>>(vb, wvb, bv, vt, nullptr, 1, 1.0f);   // V^T per head

  const dim3 gS(S_LEN / 128, S_LEN / 64, BATCH * NHEAD);        // (16, 32, 32)
  scores_gemm<<<gS, blk, 0, stream>>>(qh, kh, attn);

  softmax_rows<<<dim3(BATCH * NHEAD * S_LEN), blk, 0, stream>>>(attn, msk);

  const dim3 gAV(S_LEN / 128, BATCH * NHEAD);                   // (16, 32)
  av_gemm<<<gAV, blk, 0, stream>>>(attn, vt, ctx);

  proj_gemm<<<gProj, blk, 0, stream>>>(ctx, wob, bo, nullptr, out, 2, 1.0f);
}
